// MutiheadAttention_25718264169023
// MI455X (gfx1250) — matrix operand order repaired and checked by reading, not compiled, not run on hardware
//
#include <hip/hip_runtime.h>

// MI455X / gfx1250 — wave32, WMMA 16x16x32 f16 with f32 accumulate.

typedef __attribute__((ext_vector_type(16))) _Float16 v16h;
typedef __attribute__((ext_vector_type(8)))  _Float16 h8;
typedef __attribute__((ext_vector_type(4)))  _Float16 h4;
typedef __attribute__((ext_vector_type(8)))  float    v8f;
typedef __attribute__((ext_vector_type(4)))  float    f4;

#define B_   8
#define T_   512
#define F_   512
#define H_   8
#define DH_  64
#define ML_  1024
#define HB_  64
#define M_   4096   // B*T

__device__ __forceinline__ v8f wmma32(v16h a, v16h b, v8f c) {
  // (neg_a, A, neg_b, B, c_mod, C, reuse_a, reuse_b)
  return __builtin_amdgcn_wmma_f32_16x16x32_f16(false, a, false, b, (short)0, c,
                                                false, false);
}

// A-matrix fragment (16x32 f16, M rows striped on lanes):
// lane half hs selects K chunks {8hs..8hs+7} and {16+8hs..16+8hs+7}
__device__ __forceinline__ v16h afrag(const _Float16* p, int hs) {
  union { v16h v; h8 h[2]; } u;
  u.h[0] = *(const h8*)(p + 8 * hs);
  u.h[1] = *(const h8*)(p + 16 + 8 * hs);
  return u.v;
}
// B-matrix fragment (32x16 f16, K rows striped on lane halves):
// lane half hs selects K chunks {8hs..8hs+7} and {16+8hs..16+8hs+7}, N = lane&15 ; caller passes p = &B_col_n[0]
// where consecutive elements along K are contiguous in memory.
__device__ __forceinline__ v16h bfrag(const _Float16* p, int hs) {
  union { v16h v; h8 h[2]; } u;
  u.h[0] = *(const h8*)(p + 8 * hs);
  u.h[1] = *(const h8*)(p + 16 + 8 * hs);
  return u.v;
}

// ---------------- prep kernels ----------------

__global__ __launch_bounds__(256) void cvt_f16_v4(const float* __restrict__ s,
                                                  _Float16* __restrict__ d, int n4) {
  int i = blockIdx.x * 256 + threadIdx.x;
  if (i < n4) {
    f4 v = ((const f4*)s)[i];
    ((h4*)d)[i] = __builtin_convertvector(v, h4);
  }
}

// k_full[hb][s][d] = f16(cache_k[hb][T+s][d]), s in [0, ML-T)
__global__ __launch_bounds__(256) void shift_k(const float* __restrict__ ck,
                                               _Float16* __restrict__ kf) {
  int i  = blockIdx.x * 256 + threadIdx.x;    // 64*512*16 = 524288 vec4 elems
  int d4 = i & 15;
  int s  = (i >> 4) & 511;
  int hb = i >> 13;
  f4 v = ((const f4*)(ck + ((size_t)hb * ML_ + T_ + s) * DH_))[d4];
  ((h4*)(kf + ((size_t)hb * ML_ + s) * DH_))[d4] = __builtin_convertvector(v, h4);
}

// vT[hb][d][s] = f16(cache_v[hb][T+s][d])  (transposed for contiguous PV B-frags)
__global__ __launch_bounds__(256) void shift_v(const float* __restrict__ cv,
                                               _Float16* __restrict__ vt) {
  int i  = blockIdx.x * 256 + threadIdx.x;    // 64*512*64 = 2097152
  int d  = i & 63;
  int s  = (i >> 6) & 511;
  int hb = i >> 15;
  vt[((size_t)hb * DH_ + d) * ML_ + s] =
      (_Float16)cv[((size_t)hb * ML_ + T_ + s) * DH_ + d];
}

// ---------------- generic 4096x512x512 GEMM:  C = A @ W^T + bias ----------------
// mode 0: -> q_heads[hb][t][d] f16      mode 1: -> k_full[hb][T+t][d] f16
// mode 2: -> vT[hb][d][T+t]    f16      mode 3: -> out fp32 (B,T,F)
__global__ __launch_bounds__(256) void gemm512(const _Float16* __restrict__ A,
                                               const _Float16* __restrict__ W,
                                               const float* __restrict__ bias,
                                               float* __restrict__ outf,
                                               _Float16* __restrict__ outh, int mode) {
  int lane = threadIdx.x & 31, w = threadIdx.x >> 5;
  int hs = lane >> 4, ln = lane & 15;
  int m0 = blockIdx.x << 4;
  int n0 = (blockIdx.y << 7) + (w << 4);

  const _Float16* arow = A + (size_t)(m0 + ln) * F_;   // A row for this lane
  const _Float16* wrow = W + (size_t)(n0 + ln) * F_;   // B=W^T col n == W row n
  v8f c = {};
#pragma unroll
  for (int kk = 0; kk < F_; kk += 32)
    c = wmma32(afrag(arow + kk, hs), bfrag(wrow + kk, hs), c);

  int n = n0 + ln;
  float bvv = bias[n];
  if (mode == 3) {
#pragma unroll
    for (int r = 0; r < 8; ++r)
      outf[(size_t)(m0 + r + 8 * hs) * F_ + n] = c[r] + bvv;
  } else {
    int hh = n >> 6, dd = n & 63;
#pragma unroll
    for (int r = 0; r < 8; ++r) {
      int m = m0 + r + 8 * hs;
      int b = m >> 9, t = m & 511;
      float val = c[r] + bvv;
      size_t hb = (size_t)(hh * B_ + b);
      if (mode == 0)
        outh[(hb * T_ + t) * DH_ + dd] = (_Float16)val;
      else if (mode == 1)
        outh[(hb * ML_ + T_ + t) * DH_ + dd] = (_Float16)val;
      else
        outh[(hb * DH_ + dd) * ML_ + T_ + t] = (_Float16)val;
    }
  }
}

// ---------------- flash attention with Gaussian relative-position mask ----------------
__global__ __launch_bounds__(256) void attn512(const _Float16* __restrict__ q,
                                               const _Float16* __restrict__ kf,
                                               const _Float16* __restrict__ vT,
                                               const float* __restrict__ delta,
                                               _Float16* __restrict__ ax) {
  __shared__ __align__(16) float pbuf[8][16][36];  // per-wave P tile, padded stride
  int lane = threadIdx.x & 31, w = threadIdx.x >> 5;
  int hs = lane >> 4, ln = lane & 15;
  int hb = blockIdx.y;
  int t0 = ((blockIdx.x << 3) + w) << 4;
  int hh = hb >> 3, bb = hb & 7;

  float dl = delta[0];
  float inv_den = 1.0f / (dl * dl + 1e-8f);
  const float inv_d = 0.044194173824159216f;  // 1/sqrt(512)

  const _Float16* qrow = q + ((size_t)hb * T_ + t0 + ln) * DH_;
  v16h qa0 = afrag(qrow, hs);        // K = 0..31 of dh
  v16h qa1 = afrag(qrow + 32, hs);   // K = 32..63

  float mrow[8], lrow[8];
  v8f acc0 = {}, acc1 = {}, acc2 = {}, acc3 = {};
#pragma unroll
  for (int r = 0; r < 8; ++r) { mrow[r] = -1e30f; lrow[r] = 0.f; }

  for (int s0 = 0; s0 < ML_; s0 += 32) {
    // S = q @ k^T for 32 keys (two 16-wide key tiles)
    const _Float16* krow0 = kf + ((size_t)hb * ML_ + s0 + ln) * DH_;
    const _Float16* krow1 = krow0 + 16 * DH_;
    v8f S0 = {}, S1 = {};
    S0 = wmma32(qa0, bfrag(krow0, hs), S0);
    S0 = wmma32(qa1, bfrag(krow0 + 32, hs), S0);
    S1 = wmma32(qa0, bfrag(krow1, hs), S1);
    S1 = wmma32(qa1, bfrag(krow1 + 32, hs), S1);

    float p0[8], p1[8];
#pragma unroll
    for (int r = 0; r < 8; ++r) {
      float pq = (float)(ML_ - T_ + t0 + r + 8 * hs);
      float d0 = pq - (float)(s0 + ln);
      float d1 = pq - (float)(s0 + 16 + ln);
      float v0 = S0[r] * __expf(-d0 * d0 * inv_den) * inv_d;
      float v1 = S1[r] * __expf(-d1 * d1 * inv_den) * inv_d;
      float mx = fmaxf(v0, v1);                 // row max across 16-lane half
      mx = fmaxf(mx, __shfl_xor(mx, 1, 32));
      mx = fmaxf(mx, __shfl_xor(mx, 2, 32));
      mx = fmaxf(mx, __shfl_xor(mx, 4, 32));
      mx = fmaxf(mx, __shfl_xor(mx, 8, 32));
      float mnew  = fmaxf(mrow[r], mx);
      float scale = __expf(mrow[r] - mnew);
      v0 = __expf(v0 - mnew);
      v1 = __expf(v1 - mnew);
      float rs = v0 + v1;
      rs += __shfl_xor(rs, 1, 32);
      rs += __shfl_xor(rs, 2, 32);
      rs += __shfl_xor(rs, 4, 32);
      rs += __shfl_xor(rs, 8, 32);
      lrow[r] = lrow[r] * scale + rs;
      mrow[r] = mnew;
      p0[r] = v0; p1[r] = v1;
      acc0[r] *= scale; acc1[r] *= scale; acc2[r] *= scale; acc3[r] *= scale;
    }

    // C/D-layout P -> A-layout fragment via per-wave LDS tile
#pragma unroll
    for (int r = 0; r < 8; ++r) {
      pbuf[w][r + 8 * hs][ln]      = p0[r];
      pbuf[w][r + 8 * hs][16 + ln] = p1[r];
    }
    asm volatile("s_wait_dscnt 0" ::: "memory");
    f4 a0 = *(const f4*)&pbuf[w][ln][8 * hs];
    f4 a1 = *(const f4*)&pbuf[w][ln][8 * hs + 4];
    f4 a2 = *(const f4*)&pbuf[w][ln][16 + 8 * hs];
    f4 a3 = *(const f4*)&pbuf[w][ln][20 + 8 * hs];
    v16h pa;
#pragma unroll
    for (int i = 0; i < 4; ++i) {
      pa[i]      = (_Float16)a0[i];
      pa[4 + i]  = (_Float16)a1[i];
      pa[8 + i]  = (_Float16)a2[i];
      pa[12 + i] = (_Float16)a3[i];
    }

    // O += P @ V : vT gives contiguous K along s
    const _Float16* vbase = vT + (size_t)hb * DH_ * ML_ + s0;
    acc0 = wmma32(pa, bfrag(vbase + (size_t)(0 + ln) * ML_, hs), acc0);
    acc1 = wmma32(pa, bfrag(vbase + (size_t)(16 + ln) * ML_, hs), acc1);
    acc2 = wmma32(pa, bfrag(vbase + (size_t)(32 + ln) * ML_, hs), acc2);
    acc3 = wmma32(pa, bfrag(vbase + (size_t)(48 + ln) * ML_, hs), acc3);
  }

#pragma unroll
  for (int r = 0; r < 8; ++r) {
    float inv_l = 1.0f / lrow[r];
    int t = t0 + r + 8 * hs;
    size_t base = ((size_t)bb * T_ + t) * F_ + (size_t)hh * DH_;
    ax[base + 0  + ln] = (_Float16)(acc0[r] * inv_l);
    ax[base + 16 + ln] = (_Float16)(acc1[r] * inv_l);
    ax[base + 32 + ln] = (_Float16)(acc2[r] * inv_l);
    ax[base + 48 + ln] = (_Float16)(acc3[r] * inv_l);
  }
}

// ---------------- host ----------------

extern "C" void kernel_launch(void* const* d_in, const int* in_sizes, int n_in,
                              void* d_out, int out_size, void* d_ws, size_t ws_size,
                              hipStream_t stream) {
  const float* x     = (const float*)d_in[0];
  const float* Wq    = (const float*)d_in[1];
  const float* bq    = (const float*)d_in[2];
  const float* Wk    = (const float*)d_in[3];
  const float* bk    = (const float*)d_in[4];
  const float* Wv    = (const float*)d_in[5];
  const float* bv    = (const float*)d_in[6];
  const float* Wo    = (const float*)d_in[7];
  const float* bo    = (const float*)d_in[8];
  const float* delta = (const float*)d_in[9];
  const float* ck    = (const float*)d_in[10];
  const float* cv    = (const float*)d_in[11];
  float* out = (float*)d_out;
  (void)in_sizes; (void)n_in; (void)out_size; (void)ws_size;

  char* ws = (char*)d_ws;
  _Float16* x_h  = (_Float16*)(ws + 0);           // 4  MiB (B*T*F f16)
  _Float16* wq_h = (_Float16*)(ws + 4194304);     // 512 KiB
  _Float16* wk_h = (_Float16*)(ws + 4718592);
  _Float16* wv_h = (_Float16*)(ws + 5242880);
  _Float16* wo_h = (_Float16*)(ws + 5767168);
  _Float16* q_h  = (_Float16*)(ws + 6291456);     // 4  MiB (HB,T,dh)
  _Float16* k_h  = (_Float16*)(ws + 10485760);    // 8  MiB (HB,ML,dh)
  _Float16* vt_h = (_Float16*)(ws + 18874368);    // 8  MiB (HB,dh,ML)
  _Float16* ax_h = (_Float16*)(ws + 27262976);    // 4  MiB (B,T,F)

  // prep: f32->f16 conversions + streaming-cache shift (v transposed)
  cvt_f16_v4<<<2048, 256, 0, stream>>>(x,  x_h,  524288);
  cvt_f16_v4<<<256,  256, 0, stream>>>(Wq, wq_h, 65536);
  cvt_f16_v4<<<256,  256, 0, stream>>>(Wk, wk_h, 65536);
  cvt_f16_v4<<<256,  256, 0, stream>>>(Wv, wv_h, 65536);
  cvt_f16_v4<<<256,  256, 0, stream>>>(Wo, wo_h, 65536);
  shift_k<<<2048, 256, 0, stream>>>(ck, k_h);
  shift_v<<<8192, 256, 0, stream>>>(cv, vt_h);

  // projections (WMMA GEMMs) with fused head-major / transposed scatters
  dim3 gg(256, 4);
  gemm512<<<gg, 256, 0, stream>>>(x_h, wq_h, bq, nullptr, q_h, 0);
  gemm512<<<gg, 256, 0, stream>>>(x_h, wk_h, bk, nullptr, k_h, 1);
  gemm512<<<gg, 256, 0, stream>>>(x_h, wv_h, bv, nullptr, vt_h, 2);

  // flash attention with Gaussian relative-position mask
  dim3 ga(4, 64);
  attn512<<<ga, 256, 0, stream>>>(q_h, k_h, vt_h, delta, ax_h);

  // output projection -> fp32 result
  gemm512<<<gg, 256, 0, stream>>>(ax_h, wo_h, bo, out, nullptr, 3);
}
